// MiniPointNets_60481729462414
// MI455X (gfx1250) — matrix operand order repaired and checked by reading, not compiled, not run on hardware
//
#include <hip/hip_runtime.h>
#include <hip/hip_bf16.h>

typedef __attribute__((ext_vector_type(16))) _Float16 v16h;
typedef __attribute__((ext_vector_type(8)))  _Float16 v8h;
typedef __attribute__((ext_vector_type(8)))  float    v8f;

#define NS   128
#define NPT  8192
#define NWAVES 8

// ---- packed A-fragment table (1 frag = 32 lanes * 16 halves = 512 halves) ----
#define FRAG_PRE0 0
#define FRAG_PRE1 1
#define FRAG_PRE2 5
#define FRAG_G0   69
#define FRAG_G1   325
#define FRAG_G2   453
#define FRAG_P0   517
#define FRAG_P1   901
#define FRAG_P2   933
#define NFRAGS    935

// ======================= weight fragment packing =======================
struct LayerDesc { const float* W; int cin, cout, Kt, Mt, fragBase; };
struct PrepArgs  { LayerDesc L[9]; };

__global__ void prep_weights(PrepArgs args, _Float16* __restrict__ wpack)
{
    const int frag = blockIdx.x;
    const int lane = threadIdx.x;            // 32 threads
    int li = 0;
    for (int i = 0; i < 9; ++i) {
        int fb = args.L[i].fragBase;
        if (frag >= fb && frag < fb + args.L[i].Kt * args.L[i].Mt) { li = i; break; }
    }
    LayerDesc d = args.L[li];
    const int lf = frag - d.fragBase;
    const int mt = lf / d.Kt;
    const int kt = lf % d.Kt;
    const int m  = mt * 16 + (lane & 15);
    _Float16* out = wpack + (size_t)frag * 512 + lane * 16;
    const int khi = (lane >> 4) << 4;        // 0 or 16
#pragma unroll
    for (int e = 0; e < 16; ++e) {
        // A-fragment K mapping: lanes 0-15 -> K {0..15}; lanes 16-31 -> K {16..31}
        int k = kt * 32 + e + khi;
        float v = (m < d.cout && k < d.cin) ? d.W[(size_t)m * d.cin + k] : 0.f;
        out[e] = (_Float16)v;
    }
}

// ======================= WMMA MLP layer (LDS -> LDS) =======================
// Input:  lin  pixel-major [128][SIN] halves (channels 0..Kt*32-1 valid/zero-padded)
// Output: lout pixel-major [128][SOUT] halves, channels 0..Mt*16-1
// Strides padded (+8 halves, stride_dwords == 4 mod 8) so the 16 px-lanes of a
// B-fragment ds_load_b128 tile all 64 LDS banks conflict-free.
__device__ __forceinline__ v16h load_bfrag(const _Float16* p)
{
    v16h b;
    *((v8h*)&b)     = *(const v8h*)(p);      // ds_load_b128
    *((v8h*)&b + 1) = *(const v8h*)(p + 8);  // ds_load_b128
    return b;
}

template <int Kt, int Mt, int SIN, int SOUT, int RELU>
__device__ __forceinline__ void mlp_layer_t(
    const _Float16* __restrict__ wpack, const float* __restrict__ bias,
    const _Float16* __restrict__ lin, _Float16* __restrict__ lout, int tid)
{
    const int lane = tid & 31;
    const int wave = tid >> 5;
    const int px   = lane & 15;
    const int khi  = lane >> 4;              // 0/1 : which 16-K half this lane holds
    if (Mt >= NWAVES) {
        // Each wave owns an m-tile: one A-fragment load feeds 8 independent
        // WMMAs (all n-tiles) -> 8x less weight traffic, no D->A/B hazards.
        for (int mt = wave; mt < Mt; mt += NWAVES) {
            v8f acc[8];
#pragma unroll
            for (int nt = 0; nt < 8; ++nt)
                acc[nt] = v8f{0.f,0.f,0.f,0.f,0.f,0.f,0.f,0.f};
            const _Float16* aa = wpack + ((size_t)mt * Kt) * 512 + lane * 16;
            const _Float16* bb = lin + px * SIN + khi * 16;
            for (int kt = 0; kt < Kt; ++kt) {
                v16h a = *(const v16h*)aa;               // 2x global_load_b128
                aa += 512;
#pragma unroll
                for (int nt = 0; nt < 8; ++nt) {
                    v16h b = load_bfrag(bb + nt * 16 * SIN);
                    acc[nt] = __builtin_amdgcn_wmma_f32_16x16x32_f16(
                                  false, a, false, b, (short)0, acc[nt], false, false);
                }
                bb += 32;
            }
            const float* bp = bias + mt * 16 + khi * 8;
            float bv[8];
#pragma unroll
            for (int r = 0; r < 8; ++r) bv[r] = bp[r];
#pragma unroll
            for (int nt = 0; nt < 8; ++nt) {
                _Float16* op = lout + (nt * 16 + px) * SOUT + mt * 16 + khi * 8;
#pragma unroll
                for (int r = 0; r < 8; ++r) {
                    float v = acc[nt][r] + bv[r];
                    if (RELU) v = fmaxf(v, 0.f);
                    op[r] = (_Float16)v;
                }
            }
        }
    } else {
        // Small-M layers: frag-parallel so all 8 waves contribute.
        for (int frag = wave; frag < Mt * 8; frag += NWAVES) {
            const int mt = frag >> 3;
            const int nt = frag & 7;
            v8f acc = {0.f,0.f,0.f,0.f,0.f,0.f,0.f,0.f};
            const _Float16* aa = wpack + ((size_t)mt * Kt) * 512 + lane * 16;
            const _Float16* bb = lin + (nt * 16 + px) * SIN + khi * 16;
#pragma unroll
            for (int kt = 0; kt < Kt; ++kt) {
                v16h a = *(const v16h*)aa;
                v16h b = load_bfrag(bb);
                acc = __builtin_amdgcn_wmma_f32_16x16x32_f16(
                          false, a, false, b, (short)0, acc, false, false);
                aa += 512;
                bb += 32;
            }
            _Float16*    op = lout + (nt * 16 + px) * SOUT + mt * 16 + khi * 8;
            const float* bp = bias + mt * 16 + khi * 8;
#pragma unroll
            for (int r = 0; r < 8; ++r) {
                float v = acc[r] + bp[r];
                if (RELU) v = fmaxf(v, 0.f);
                op[r] = (_Float16)v;
            }
        }
    }
}

// ======================= gathered point data + U12 =======================
// PT float layout: X0 0, X1 384, F0 768, F1 1152, T0 1536, T1 1920, R0 2304, R1 3456 (4608 floats)
__device__ __forceinline__ void load_points(
    float* __restrict__ PT, int b,
    const int* __restrict__ sub, const float* __restrict__ xyz,
    const float* __restrict__ flow, const float* __restrict__ arot,
    const float* __restrict__ atr, int tid)
{
    if (tid < NS) {
        const int j  = tid;
        const int i0 = sub[(b * 2 + 0) * NS + j];
        const int i1 = sub[(b * 2 + 1) * NS + j];
        const size_t p0 = (size_t)(b * 2 + 0) * NPT + i0;
        const size_t p1 = (size_t)(b * 2 + 1) * NPT + i1;
#pragma unroll
        for (int d = 0; d < 3; ++d) {
            PT[0   + j*3 + d] = xyz [p0*3 + d];
            PT[384 + j*3 + d] = xyz [p1*3 + d];
            PT[768 + j*3 + d] = flow[p0*3 + d];
            PT[1152+ j*3 + d] = flow[p1*3 + d];
            PT[1536+ j*3 + d] = atr [p0*3 + d];
            PT[1920+ j*3 + d] = atr [p1*3 + d];
        }
#pragma unroll
        for (int q = 0; q < 9; ++q) {
            PT[2304 + j*9 + q] = arot[p0*9 + q];
            PT[3456 + j*9 + q] = arot[p1*9 + q];
        }
    }
}

__device__ __forceinline__ void residual3(
    const float* __restrict__ R, const float* a, const float* s,
    const float* fsub, float* r)
{
#pragma unroll
    for (int k = 0; k < 3; ++k) {
        float acc = -fsub[k] - s[k];
#pragma unroll
        for (int d = 0; d < 3; ++d) acc += (a[d] - s[d]) * R[d * 3 + k];
        r[k] = acc;
    }
}

__device__ __forceinline__ void build_u12(
    const float* __restrict__ PT, _Float16* __restrict__ act0,
    _Float16* __restrict__ act1, int b, int fixed, int colpass,
    float* __restrict__ res0o, float* __restrict__ res1o, int writeRes, int tid)
{
    if (tid >= NS) return;
    const int other = tid;
    const int n = colpass ? other : fixed;
    const int m = colpass ? fixed : other;
    const float* X0 = PT,        *X1 = PT + 384;
    const float* F0 = PT + 768,  *F1 = PT + 1152;
    const float* T0 = PT + 1536, *T1 = PT + 1920;
    const float* R0 = PT + 2304, *R1 = PT + 3456;

    float a[3], s[3], r0[3], r1[3], u3[3];
    // res0_orig[n,m,k]
#pragma unroll
    for (int d = 0; d < 3; ++d) {
        a[d] = X1[m*3+d] - X0[n*3+d];
        s[d] = F1[m*3+d] + T0[n*3+d] + F0[n*3+d];
    }
    residual3(R0 + n*9, a, s, F1 + m*3, r0);
    // res1_orig[n,m,k]
#pragma unroll
    for (int d = 0; d < 3; ++d) {
        a[d] = X0[m*3+d] - X1[n*3+d];
        s[d] = F0[m*3+d] + T1[n*3+d] + F1[n*3+d];
    }
    residual3(R1 + n*9, a, s, F0 + m*3, r1);
    // u3..5 = res1_orig[m,n,k]  (swapaxes(-1,-2) of transposed res1)
#pragma unroll
    for (int d = 0; d < 3; ++d) {
        a[d] = X0[n*3+d] - X1[m*3+d];
        s[d] = F0[n*3+d] + T1[m*3+d] + F1[m*3+d];
    }
    residual3(R1 + m*9, a, s, F0 + n*3, u3);

    if (writeRes) {
#pragma unroll
        for (int k = 0; k < 3; ++k) {
            res0o[(((size_t)b*3 + k)*NS + n)*NS + m] = r0[k];
            res1o[(((size_t)b*3 + k)*NS + n)*NS + m] = r1[k];
        }
    }
    _Float16* u = act0 + other * 40;
#pragma unroll
    for (int k = 0; k < 3; ++k) {
        u[k]   = (_Float16)r0[k];
        u[3+k] = (_Float16)u3[k];
        u[6+k] = (_Float16)X0[n*3+k];
        u[9+k] = (_Float16)X1[m*3+k];
    }
    for (int c = 12; c < 32; ++c) u[c] = (_Float16)0.f;     // K-pad for layer0
    _Float16* a1 = act1 + other * 40;
    for (int c = 16; c < 32; ++c) a1[c] = (_Float16)0.f;    // K-pad for layer1
}

// ======================= pass 1: pre-MLP + row/col max =======================
// smem: PT[0,18432) act0[18432,28672) act1[28672,38912) act2[38912,57344) big[57344,190464)
__global__ void __launch_bounds__(256) pre_max_kernel(
    const float* __restrict__ xyz, const float* __restrict__ flow,
    const int* __restrict__ sub, const float* __restrict__ arot,
    const float* __restrict__ atr, const _Float16* __restrict__ wpack,
    const float* __restrict__ pb0, const float* __restrict__ pb1,
    const float* __restrict__ pb2,
    float* __restrict__ maxbuf, float* __restrict__ res0o,
    float* __restrict__ res1o, int colpass)
{
    extern __shared__ char smem[];
    float*    PT   = (float*)smem;
    _Float16* act0 = (_Float16*)(smem + 18432);
    _Float16* act1 = (_Float16*)(smem + 28672);
    _Float16* act2 = (_Float16*)(smem + 38912);
    _Float16* big  = (_Float16*)(smem + 57344);   // [128][520]
    const int tid = threadIdx.x;
    const int wg  = blockIdx.x;
    const int b = wg >> 7, fixed = wg & 127;

    load_points(PT, b, sub, xyz, flow, arot, atr, tid);
    __syncthreads();
    build_u12(PT, act0, act1, b, fixed, colpass, res0o, res1o, colpass == 0, tid);
    __syncthreads();
    mlp_layer_t<1, 1, 40, 40, 1>(wpack + (size_t)FRAG_PRE0*512, pb0, act0, act1, tid); __syncthreads();
    mlp_layer_t<1, 4, 40, 72, 1>(wpack + (size_t)FRAG_PRE1*512, pb1, act1, act2, tid); __syncthreads();
    mlp_layer_t<2, 32, 72, 520, 1>(wpack + (size_t)FRAG_PRE2*512, pb2, act2, big, tid); __syncthreads();
    // max over the 128 local pixels (ReLU output >= 0, so 0-init is the identity)
    for (int c = tid; c < 512; c += 256) {
        float mx = 0.f;
        for (int p = 0; p < 128; ++p) mx = fmaxf(mx, (float)big[p * 520 + c]);
        maxbuf[(((size_t)b*2 + colpass)*NS + fixed)*512 + c] = mx;
    }
}

// ======================= pass 2: glob MLP on 2048 pooled vectors =======================
// smem: in520[0,133120) a264[133120,200704) b264[200704,268288) o136 reuses [0,...)
__global__ void __launch_bounds__(256) glob_kernel(
    const float* __restrict__ maxbuf, const _Float16* __restrict__ wpack,
    const float* __restrict__ gb0, const float* __restrict__ gb1,
    const float* __restrict__ gb2, _Float16* __restrict__ gbuf)
{
    extern __shared__ char smem[];
    _Float16* in520 = (_Float16*)smem;
    _Float16* a264  = (_Float16*)(smem + 133120);
    _Float16* b264  = (_Float16*)(smem + 200704);
    _Float16* o136  = (_Float16*)smem;            // in520 dead after G0
    const int tid = threadIdx.x;
    const size_t vbase = (size_t)blockIdx.x * 128;
    for (int idx = tid; idx < 128 * 512; idx += 256) {
        int p = idx >> 9, c = idx & 511;
        in520[p * 520 + c] = (_Float16)maxbuf[(vbase + p) * 512 + c];
    }
    __syncthreads();
    mlp_layer_t<16, 16, 520, 264, 1>(wpack + (size_t)FRAG_G0*512, gb0, in520, a264, tid); __syncthreads();
    mlp_layer_t<8, 16, 264, 264, 1>(wpack + (size_t)FRAG_G1*512, gb1, a264, b264, tid); __syncthreads();
    mlp_layer_t<8, 8, 264, 136, 1>(wpack + (size_t)FRAG_G2*512, gb2, b264, o136, tid); __syncthreads();
    for (int idx = tid; idx < 128 * 128; idx += 256) {
        int p = idx >> 7, c = idx & 127;
        gbuf[(vbase + p) * 128 + c] = o136[p * 136 + c];
    }
}

// ======================= pass 3: recompute pre + post MLP =======================
// smem: PT/act0/act1/act2 as pass1; big768[67584,266240); out256 reuses [0,67584);
// out64 reuses [67584,86016); out16 at [86016,96256)
__global__ void __launch_bounds__(256) post_kernel(
    const float* __restrict__ xyz, const float* __restrict__ flow,
    const int* __restrict__ sub, const float* __restrict__ arot,
    const float* __restrict__ atr, const _Float16* __restrict__ wpack,
    const float* __restrict__ pb0, const float* __restrict__ pb1,
    const float* __restrict__ pb2,
    const float* __restrict__ qb0, const float* __restrict__ qb1,
    const float* __restrict__ qb2,
    const float* __restrict__ w3, const float* __restrict__ b3,
    const _Float16* __restrict__ gbuf, float* __restrict__ outU)
{
    extern __shared__ char smem[];
    float*    PT     = (float*)smem;
    _Float16* act0   = (_Float16*)(smem + 18432);
    _Float16* act1   = (_Float16*)(smem + 28672);
    _Float16* act2   = (_Float16*)(smem + 38912);
    _Float16* big    = (_Float16*)(smem + 67584);   // [128][776]
    _Float16* out256 = (_Float16*)smem;             // [128][264], pre scratch dead
    _Float16* out64  = (_Float16*)(smem + 67584);   // [128][72], big dead after P0
    _Float16* out16  = (_Float16*)(smem + 86016);   // [128][40]
    const int tid = threadIdx.x;
    const int wg  = blockIdx.x;
    const int b = wg >> 7, n = wg & 127;

    load_points(PT, b, sub, xyz, flow, arot, atr, tid);
    __syncthreads();
    build_u12(PT, act0, act1, b, n, 0, nullptr, nullptr, 0, tid);
    __syncthreads();
    mlp_layer_t<1, 1, 40, 40, 1>(wpack + (size_t)FRAG_PRE0*512, pb0, act0, act1, tid); __syncthreads();
    mlp_layer_t<1, 4, 40, 72, 1>(wpack + (size_t)FRAG_PRE1*512, pb1, act1, act2, tid); __syncthreads();
    mlp_layer_t<2, 32, 72, 776, 1>(wpack + (size_t)FRAG_PRE2*512, pb2, act2, big, tid); __syncthreads();
    // concat channels 512..767 : g0[b][n][:] (broadcast over m), g1[b][m][:]
    {
        const _Float16* g0p = gbuf + ((size_t)(b*2 + 0)*NS + n) * 128;
        const _Float16* g1p = gbuf + ((size_t)(b*2 + 1)*NS) * 128;
        for (int idx = tid; idx < 128 * 256; idx += 256) {
            int p = idx >> 8, c = idx & 255;
            big[p * 776 + 512 + c] = (c < 128) ? g0p[c] : g1p[p * 128 + (c - 128)];
        }
    }
    __syncthreads();
    mlp_layer_t<24, 16, 776, 264, 1>(wpack + (size_t)FRAG_P0*512, qb0, big, out256, tid); __syncthreads();
    mlp_layer_t<8, 4, 264, 72, 1>(wpack + (size_t)FRAG_P1*512, qb1, out256, out64, tid); __syncthreads();
    mlp_layer_t<2, 1, 72, 40, 1>(wpack + (size_t)FRAG_P2*512, qb2, out64, out16, tid); __syncthreads();
    // final 16 -> 1 (no ReLU) in VALU
    if (tid < NS) {
        const int m = tid;
        float acc = b3[0];
#pragma unroll
        for (int c = 0; c < 16; ++c) acc += (float)out16[m * 40 + c] * w3[c];
        outU[((size_t)b * NS + n) * NS + m] = acc;
    }
}

// ======================= host launcher =======================
extern "C" void kernel_launch(void* const* d_in, const int* in_sizes, int n_in,
                              void* d_out, int out_size, void* d_ws, size_t ws_size,
                              hipStream_t stream)
{
    (void)out_size; (void)ws_size;
    const float* xyz  = (const float*)d_in[0];
    const float* flow = (const float*)d_in[1];
    const int*   sub  = (const int*)d_in[2];
    const float* arot = (const float*)d_in[3];
    const float* atr  = (const float*)d_in[4];

    const float *preW[3], *preB[3], *globW[3], *globB[3], *postW[4], *postB[4];
    if (n_in >= 25 && in_sizes[5] == 192) {
        // params flattened in insertion order: pre(3), glob(3), post(4)
        int i = 5;
        for (int l = 0; l < 3; ++l) { preW[l]  = (const float*)d_in[i++]; preB[l]  = (const float*)d_in[i++]; }
        for (int l = 0; l < 3; ++l) { globW[l] = (const float*)d_in[i++]; globB[l] = (const float*)d_in[i++]; }
        for (int l = 0; l < 4; ++l) { postW[l] = (const float*)d_in[i++]; postB[l] = (const float*)d_in[i++]; }
    } else {
        // sorted-key flatten order: glob, post, pre
        int i = 5;
        for (int l = 0; l < 3; ++l) { globW[l] = (const float*)d_in[i++]; globB[l] = (const float*)d_in[i++]; }
        for (int l = 0; l < 4; ++l) { postW[l] = (const float*)d_in[i++]; postB[l] = (const float*)d_in[i++]; }
        for (int l = 0; l < 3; ++l) { preW[l]  = (const float*)d_in[i++]; preB[l]  = (const float*)d_in[i++]; }
    }

    // workspace: [0,1MB) packed weights, [1MB,5MB) maxbuf f32, [5MB,5.5MB) gbuf f16
    _Float16* wpack  = (_Float16*)d_ws;
    float*    maxbuf = (float*)((char*)d_ws + (1u << 20));
    _Float16* gbuf   = (_Float16*)((char*)d_ws + (1u << 20) + 2048u * 512u * 4u);

    float* outU  = (float*)d_out;
    float* res0o = outU + 8 * 128 * 128;
    float* res1o = res0o + 8 * 3 * 128 * 128;

    PrepArgs pa;
    {
        const int cins [9] = {12, 16, 64,  512, 256, 256,  768, 256, 64};
        const int couts[9] = {16, 64, 512, 256, 256, 128,  256, 64,  16};
        const int Kts  [9] = {1, 1, 2,     16, 8, 8,       24, 8, 2};
        const int Mts  [9] = {1, 4, 32,    16, 16, 8,      16, 4, 1};
        const int bases[9] = {FRAG_PRE0, FRAG_PRE1, FRAG_PRE2, FRAG_G0, FRAG_G1,
                              FRAG_G2, FRAG_P0, FRAG_P1, FRAG_P2};
        const float* Ws[9] = {preW[0], preW[1], preW[2], globW[0], globW[1],
                              globW[2], postW[0], postW[1], postW[2]};
        for (int i = 0; i < 9; ++i)
            pa.L[i] = LayerDesc{Ws[i], cins[i], couts[i], Kts[i], Mts[i], bases[i]};
    }

    const int SM1 = 190464;   // pre/max pass
    const int SM2 = 268288;   // glob pass
    const int SM3 = 266240;   // post pass
    (void)hipFuncSetAttribute((const void*)pre_max_kernel,
                              hipFuncAttributeMaxDynamicSharedMemorySize, SM1);
    (void)hipFuncSetAttribute((const void*)glob_kernel,
                              hipFuncAttributeMaxDynamicSharedMemorySize, SM2);
    (void)hipFuncSetAttribute((const void*)post_kernel,
                              hipFuncAttributeMaxDynamicSharedMemorySize, SM3);

    prep_weights<<<NFRAGS, 32, 0, stream>>>(pa, wpack);

    pre_max_kernel<<<8 * NS, 256, SM1, stream>>>(xyz, flow, sub, arot, atr, wpack,
        preB[0], preB[1], preB[2], maxbuf, res0o, res1o, 0);
    pre_max_kernel<<<8 * NS, 256, SM1, stream>>>(xyz, flow, sub, arot, atr, wpack,
        preB[0], preB[1], preB[2], maxbuf, res0o, res1o, 1);

    glob_kernel<<<16, 256, SM2, stream>>>(maxbuf, wpack, globB[0], globB[1], globB[2], gbuf);

    post_kernel<<<8 * NS, 256, SM3, stream>>>(xyz, flow, sub, arot, atr, wpack,
        preB[0], preB[1], preB[2], postB[0], postB[1], postB[2],
        postW[3], postB[3], gbuf, outU);
}
